// ScheduledSamplingPredictor_81252191305751
// MI455X (gfx1250) — compile-verified, matrix operand order checked by reading, not run on hardware
//
#include <hip/hip_runtime.h>

// Problem constants (from reference)
#define L_DIM 128
#define H_DIM 1536           // LSTM units
#define K_DIM 1792           // 2L + H  (fused [x || h] input width)
#define B_DIM 256            // batch
#define S_DIM 200            // sequence length
#define FOURH (4 * H_DIM)    // 6144
#define OUT_DIM 256          // 2L
#define KSTEP 32
#define NIT (K_DIM / KSTEP)  // 56 K-iterations in the LSTM GEMM

#define AS1 __attribute__((address_space(1)))
#define AS3 __attribute__((address_space(3)))

#ifndef __has_builtin
#define __has_builtin(x) 0
#endif
#if __has_builtin(__builtin_amdgcn_global_load_async_to_lds_b128)
#define HAVE_ASYNC_LDS 1
#else
#define HAVE_ASYNC_LDS 0
#endif

typedef __attribute__((ext_vector_type(16))) __bf16 v16bf;
typedef __attribute__((ext_vector_type(8)))  float  v8f;
typedef __attribute__((ext_vector_type(4)))  int    v4i;

union AFrag { v16bf v; uint4 q[2]; };

static __device__ __forceinline__ __bf16 f2bf(float f) {
  // round-to-nearest-even f32 -> bf16
  unsigned u = __builtin_bit_cast(unsigned, f);
  unsigned r = u + 0x7FFFu + ((u >> 16) & 1u);
  unsigned short s = (unsigned short)(r >> 16);
  return __builtin_bit_cast(__bf16, s);
}

static __device__ __forceinline__ float sigmoidf(float x) {
  return 1.0f / (1.0f + __expf(-x));
}

// ---------------------------------------------------------------------------
// One-time (per launch) weight prep: W^T bf16 with gate-major layout.
// Wt[g][n][k] = Wfused[k][g*H + n], where Wfused = [kernel ; rec_kernel].
// ---------------------------------------------------------------------------
__global__ __launch_bounds__(256) void pack_w_kernel(
    const float* __restrict__ kern, const float* __restrict__ rec,
    __bf16* __restrict__ Wt) {
  const int total = 4 * H_DIM * K_DIM;
  for (int idx = blockIdx.x * blockDim.x + threadIdx.x; idx < total;
       idx += gridDim.x * blockDim.x) {
    int k   = idx % K_DIM;
    int col = idx / K_DIM;  // == g*H + n, exactly the fused-weight column
    float v = (k < 2 * L_DIM) ? kern[(size_t)k * FOURH + col]
                              : rec[(size_t)(k - 2 * L_DIM) * FOURH + col];
    Wt[idx] = f2bf(v);
  }
}

// WoT[n][k] = w_out[k][n]   (n in [0,256), k in [0,1536))
__global__ __launch_bounds__(256) void pack_wo_kernel(
    const float* __restrict__ w_out, __bf16* __restrict__ WoT) {
  const int total = OUT_DIM * H_DIM;
  for (int idx = blockIdx.x * blockDim.x + threadIdx.x; idx < total;
       idx += gridDim.x * blockDim.x) {
    int n = idx / H_DIM;
    int k = idx % H_DIM;
    WoT[idx] = f2bf(w_out[(size_t)k * OUT_DIM + n]);
  }
}

// Init: Xh0 = [bf16(z_means[:,0]) || bf16(z_log_vars[:,0]) || 0...], C = 0.
__global__ __launch_bounds__(256) void init_kernel(
    const float* __restrict__ zm, const float* __restrict__ lv,
    __bf16* __restrict__ Xh0, float* __restrict__ Cst) {
  const int nxh = B_DIM * K_DIM;
  const int total = nxh + B_DIM * H_DIM;
  for (int idx = blockIdx.x * blockDim.x + threadIdx.x; idx < total;
       idx += gridDim.x * blockDim.x) {
    if (idx < nxh) {
      int m = idx / K_DIM, col = idx % K_DIM;
      float v = 0.0f;
      if (col < L_DIM)            v = zm[(size_t)m * (S_DIM * L_DIM) + col];
      else if (col < 2 * L_DIM)   v = lv[(size_t)m * (S_DIM * L_DIM) + col - L_DIM];
      Xh0[idx] = f2bf(v);
    } else {
      Cst[idx - nxh] = 0.0f;
    }
  }
}

// ---------------------------------------------------------------------------
// Fused LSTM step with LDS-staged, double-buffered B tiles.
//   Z[g] = [x||h] @ W[g] + bias[g]  (g = i,f,c~,o), gate math in the epilogue.
// Block = 128 threads (4 waves). Block tile: M=64 (4 waves x 16), N=32 over H,
// all 4 gates. Grid = 192 blocks = (16 M-tiles / 4) x 48 N-tiles.
// Per K-step the block stages 4 gates x 32 rows x 32 K of W^T (8 KB) in LDS,
// asynchronously (GLOBAL_LOAD_ASYNC_TO_LDS_B128, ASYNCcnt-tracked) while
// WMMAs consume the previous buffer via ds_load_b128.
// ---------------------------------------------------------------------------
__global__ __launch_bounds__(128) void lstm_step_kernel(
    const __bf16* __restrict__ Xh,    // [256][1792] bf16, row-major
    const __bf16* __restrict__ Wt,    // [4][1536][1792] bf16 (W^T, gate-major)
    const float*  __restrict__ bias,  // [6144] f32
    float*        __restrict__ Cst,   // [256][1536] f32 cell state (RMW)
    __bf16*       __restrict__ XhNext)// [256][1792]; h written at cols 256..
{
  __shared__ __bf16 Bsm[2][128 * KSTEP];  // 2 x 8 KB double buffer

  const int tid  = threadIdx.x;
  const int lane = tid & 31;
  const int wid  = tid >> 5;                  // 0..3
  const int nIdx = blockIdx.x >> 2;           // 0..47
  const int mq   = blockIdx.x & 3;            // 0..3
  const int m0   = (mq * 4 + wid) * 16;       // wave's 16 M rows
  const int n0   = nIdx * 32;                 // block's 32 N cols (over H)
  const int lm   = lane & 15;
  const int kh   = lane >> 4;                 // K-half select (WMMA layout)

  // ---- cooperative fill bookkeeping: 512 x 16B chunks, 4 per thread ----
  const char* gsrc[4];
  unsigned    ldst[4];
#pragma unroll
  for (int i = 0; i < 4; ++i) {
    int c    = tid + 128 * i;       // chunk id 0..511
    int row  = c >> 2;              // 0..127 : gate*32 + n-within
    int sub  = c & 3;               // 16B chunk within the 64B row
    int gate = row >> 5;
    int nw   = row & 31;
    gsrc[i] = (const char*)Wt +
              ((size_t)(gate * H_DIM + n0 + nw) * K_DIM + sub * 8) * 2;
    ldst[i] = (unsigned)(row * 64 + sub * 16);
  }

  auto fill = [&](int par, int kk) {
    char* lbase = (char*)(&Bsm[0][0]) + par * (128 * KSTEP * 2);
#pragma unroll
    for (int i = 0; i < 4; ++i) {
      const char* g = gsrc[i] + kk * 2;
      char*       l = lbase + ldst[i];
#if HAVE_ASYNC_LDS
      __builtin_amdgcn_global_load_async_to_lds_b128(
          (AS1 v4i*)(AS1 void*)const_cast<char*>(g),
          (AS3 v4i*)(AS3 void*)l, 0, 0);
#else
      *(uint4*)l = *(const uint4*)g;
#endif
    }
  };

  auto fill_sync = [&]() {
#if HAVE_ASYNC_LDS
    asm volatile("s_wait_asynccnt 0x0" ::: "memory");
#endif
    __syncthreads();
  };

  // ---- accumulators: bias folded in (all 8 acc lanes of a VGPR share N) ----
  v8f acc[4][2];
#pragma unroll
  for (int g = 0; g < 4; ++g)
#pragma unroll
    for (int t = 0; t < 2; ++t) {
      float b = bias[g * H_DIM + n0 + 16 * t + lm];
#pragma unroll
      for (int j = 0; j < 8; ++j) acc[g][t][j] = b;
    }

  // LDS fragment source offsets (elements) per (gate, t)
  unsigned bRow[4][2];
#pragma unroll
  for (int g = 0; g < 4; ++g)
#pragma unroll
    for (int t = 0; t < 2; ++t)
      bRow[g][t] = (unsigned)((g * 32 + 16 * t + lm) * KSTEP + kh * 8);

  const __bf16* arow = Xh + (size_t)(m0 + lm) * K_DIM + kh * 8;

  fill(0, 0);
  fill_sync();

  for (int it = 0; it < NIT; ++it) {
    const int par = it & 1;
    if (it + 1 < NIT) fill(par ^ 1, (it + 1) * KSTEP);  // async prefetch

    AFrag af;
    const uint4* ap = reinterpret_cast<const uint4*>(arow + it * KSTEP);
    af.q[0] = ap[0];        // K cols  kk + kh*8      .. +8
    af.q[1] = ap[2];        // K cols  kk + 16 + kh*8 .. +8

    const __bf16* lbuf = &Bsm[0][0] + par * (128 * KSTEP);
#pragma unroll
    for (int g = 0; g < 4; ++g) {
#pragma unroll
      for (int t = 0; t < 2; ++t) {
        const __bf16* bsrc = lbuf + bRow[g][t];
        AFrag bf;
        bf.q[0] = *reinterpret_cast<const uint4*>(bsrc);       // ds_load_b128
        bf.q[1] = *reinterpret_cast<const uint4*>(bsrc + 16);  // ds_load_b128
        acc[g][t] = __builtin_amdgcn_wmma_f32_16x16x32_bf16(
            false, af.v, false, bf.v, (short)0, acc[g][t], false, false);
      }
    }
    fill_sync();  // next buffer landed + everyone done reading current
  }

  // Gate epilogue: c' = sig(f)*c + sig(i)*relu(g); h = sig(o)*relu(c')
#pragma unroll
  for (int t = 0; t < 2; ++t) {
    const int n = n0 + 16 * t + lm;
#pragma unroll
    for (int j = 0; j < 8; ++j) {
      const int m = m0 + j + 8 * kh;
      float ig = sigmoidf(acc[0][t][j]);
      float fg = sigmoidf(acc[1][t][j]);
      float gg = acc[2][t][j] > 0.0f ? acc[2][t][j] : 0.0f;
      float og = sigmoidf(acc[3][t][j]);
      float cold = Cst[(size_t)m * H_DIM + n];
      float cnew = fg * cold + ig * gg;
      Cst[(size_t)m * H_DIM + n] = cnew;
      float h = og * (cnew > 0.0f ? cnew : 0.0f);
      XhNext[(size_t)m * K_DIM + 2 * L_DIM + n] = f2bf(h);
    }
  }
}

// ---------------------------------------------------------------------------
// Output projection: p = h @ w_out + b_out. Writes p (bf16) into cols 0..255
// of the activation buffer (next step's x) and, when tOut>=0, f32 to d_out.
// Grid: 16 blocks x 8 waves = 16 Mtiles x 8 Ntiles (N=256). Only 0.2 GFLOP.
// ---------------------------------------------------------------------------
__global__ __launch_bounds__(256) void out_proj_kernel(
    const __bf16* __restrict__ Xh,    // rows [K_DIM]; h lives at cols 256..
    const __bf16* __restrict__ WoT,   // [256][1536] bf16 (w_out^T)
    const float*  __restrict__ b_out, // [256]
    __bf16*       __restrict__ XhP,   // p written at cols 0..255
    float*        __restrict__ out, int tOut) {
  const int lane = threadIdx.x & 31;
  const int wid  = threadIdx.x >> 5;
  const int w    = blockIdx.x * 8 + wid;
  const int m0   = (w & 15) * 16;
  const int n0   = (w >> 4) * 32;   // 8 N tiles
  const int lm   = lane & 15;
  const int kh   = lane >> 4;

  v8f acc[2];
#pragma unroll
  for (int t = 0; t < 2; ++t) {
    float b = b_out[n0 + 16 * t + lm];
#pragma unroll
    for (int j = 0; j < 8; ++j) acc[t][j] = b;
  }

  const __bf16* arow = Xh + (size_t)(m0 + lm) * K_DIM + 2 * L_DIM + kh * 8;

  for (int kk = 0; kk < H_DIM; kk += KSTEP) {
    AFrag af;
    const uint4* ap = reinterpret_cast<const uint4*>(arow + kk);
    af.q[0] = ap[0];
    af.q[1] = ap[2];
#pragma unroll
    for (int t = 0; t < 2; ++t) {
      const __bf16* brow =
          WoT + (size_t)(n0 + 16 * t + lm) * H_DIM + kh * 8 + kk;
      AFrag bf;
      const uint4* bp = reinterpret_cast<const uint4*>(brow);
      bf.q[0] = bp[0];
      bf.q[1] = bp[2];
      acc[t] = __builtin_amdgcn_wmma_f32_16x16x32_bf16(
          false, af.v, false, bf.v, (short)0, acc[t], false, false);
    }
  }

#pragma unroll
  for (int t = 0; t < 2; ++t) {
    const int n = n0 + 16 * t + lm;
#pragma unroll
    for (int j = 0; j < 8; ++j) {
      const int m = m0 + j + 8 * kh;
      float p = acc[t][j];
      XhP[(size_t)m * K_DIM + n] = f2bf(p);
      if (tOut >= 0)
        out[((size_t)m * (S_DIM - 1) + tOut) * OUT_DIM + n] = p;
    }
  }
}

// ---------------------------------------------------------------------------
extern "C" void kernel_launch(void* const* d_in, const int* in_sizes, int n_in,
                              void* d_out, int out_size, void* d_ws,
                              size_t ws_size, hipStream_t stream) {
  const float* zm    = (const float*)d_in[0];  // z_means    [256,200,128]
  const float* lv    = (const float*)d_in[1];  // z_log_vars [256,200,128]
  const float* kern  = (const float*)d_in[2];  // kernel     [256,6144]
  const float* rec   = (const float*)d_in[3];  // rec_kernel [1536,6144]
  const float* bias  = (const float*)d_in[4];  // bias       [6144]
  const float* w_out = (const float*)d_in[5];  // w_out      [1536,256]
  const float* b_out = (const float*)d_in[6];  // b_out      [256]
  float* out = (float*)d_out;                  // [256,199,256] f32

  // Workspace carve-up (all 256B-aligned slices)
  char* ws = (char*)d_ws;
  __bf16* Wt  = (__bf16*)ws; ws += (size_t)4 * H_DIM * K_DIM * 2;   // 22.0 MB
  __bf16* WoT = (__bf16*)ws; ws += (size_t)OUT_DIM * H_DIM * 2;     // 0.75 MB
  __bf16* Xh0 = (__bf16*)ws; ws += (size_t)B_DIM * K_DIM * 2;       // 0.88 MB
  __bf16* Xh1 = (__bf16*)ws; ws += (size_t)B_DIM * K_DIM * 2;       // 0.88 MB
  float*  Cst = (float*)ws;                                         // 1.5 MB
  __bf16* Xh[2] = {Xh0, Xh1};

  // Weight prep (bf16 weights ~23 MB -> resident in the 192 MB L2; the 200
  // dependent steps then stream weights from L2, not HBM).
  pack_w_kernel<<<2048, 256, 0, stream>>>(kern, rec, Wt);
  pack_wo_kernel<<<512, 256, 0, stream>>>(w_out, WoT);
  init_kernel<<<1024, 256, 0, stream>>>(zm, lv, Xh[0], Cst);

  // 200 dependent steps; t=0 consumes ground-truth x0, its prediction is only
  // fed forward (not emitted), matching the reference exactly.
  for (int t = 0; t < S_DIM; ++t) {
    const __bf16* xin = Xh[t & 1];
    __bf16* xnx = Xh[(t + 1) & 1];
    lstm_step_kernel<<<192, 128, 0, stream>>>(xin, Wt, bias, Cst, xnx);
    out_proj_kernel<<<16, 256, 0, stream>>>(xnx, WoT, b_out, xnx, out, t - 1);
  }
}
